// SemiSupervisedLearning_66537633349974
// MI455X (gfx1250) — matrix operand order repaired and checked by reading, not compiled, not run on hardware
//
#include <hip/hip_runtime.h>

// ---------------------------------------------------------------------------
// Problem constants (from reference)
// ---------------------------------------------------------------------------
#define DIM_X   1024
#define H1DIM   2048
#define H2DIM   512
#define SLN     8192
#define SUN     16384
#define MTOT    (SLN + SUN)       // 24576
#define KTRIES  8

// ---------------------------------------------------------------------------
// Vector types for WMMA fragments (wave32, gfx1250)
// ---------------------------------------------------------------------------
typedef __bf16 v16bf __attribute__((ext_vector_type(16)));
typedef __bf16 v8bf  __attribute__((ext_vector_type(8)));
typedef __bf16 v4bf  __attribute__((ext_vector_type(4)));
typedef float  v8f   __attribute__((ext_vector_type(8)));

// ---------------------------------------------------------------------------
// Conversion kernels: f32 -> bf16 (X concat, and transposed weights)
// ---------------------------------------------------------------------------
__global__ __launch_bounds__(256) void k_cvt_x(const float* __restrict__ Xl,
                                               const float* __restrict__ Xu,
                                               __bf16* __restrict__ Xbf) {
    size_t tid = (size_t)blockIdx.x * 256 + threadIdx.x;   // one float4 per thread
    size_t i   = tid * 4;
    const size_t nl = (size_t)SLN * DIM_X;                 // divisible by 4
    float4 v;
    if (i < nl) v = ((const float4*)Xl)[tid];
    else        v = ((const float4*)Xu)[tid - nl / 4];
    v4bf o;
    o[0] = (__bf16)v.x; o[1] = (__bf16)v.y; o[2] = (__bf16)v.z; o[3] = (__bf16)v.w;
    *(v4bf*)(Xbf + i) = o;
}

// WT[n*K + k] = (bf16) W[k*N + n]   (W is K x N row-major)
__global__ __launch_bounds__(256) void k_cvt_wT(const float* __restrict__ W,
                                                __bf16* __restrict__ WT,
                                                int K, int N) {
    size_t tid = (size_t)blockIdx.x * 256 + threadIdx.x;
    int n = (int)(tid / K);
    int k = (int)(tid % K);
    WT[tid] = (__bf16)W[(size_t)k * N + n];
}

// ---------------------------------------------------------------------------
// bf16 WMMA GEMM with fused bias + ReLU, software-pipelined (K unrolled by 64,
// two register fragment buffers ping-pong so global_load_b128 for step k+32
// overlap the v_wmma ops of step k).
//   D = relu(A[M,K] @ B[K,N] + bias[N]),  B given transposed: BT[N,K]
// Block = 256 threads (8 waves), block tile 128x128, wave tile 64x32
// (waves 2(M) x 4(N); each wave holds 4x2 accumulators of 16x16).
// MODE 0: store bf16 -> outBf.
// MODE 1: store f32; rows < splitRow -> outA, rows >= splitRow -> outB.
//         splitRow is a multiple of 128 => the select is block-uniform.
// Requires M%128==0, N%128==0, K%64==0 (true for all layers here).
// ---------------------------------------------------------------------------
template <int MODE>
__global__ __launch_bounds__(256) void k_gemm_bf16(const __bf16* __restrict__ A,
                                                   const __bf16* __restrict__ BT,
                                                   const float*  __restrict__ bias,
                                                   int M, int N, int K,
                                                   __bf16* __restrict__ outBf,
                                                   float*  __restrict__ outA,
                                                   float*  __restrict__ outB,
                                                   int splitRow) {
    const int lane = threadIdx.x & 31;
    const int wave = threadIdx.x >> 5;
    const int wm   = wave & 1;            // 0..1  (M direction)
    const int wn   = wave >> 1;           // 0..3  (N direction)
    const int wm0  = blockIdx.x * 128 + wm * 64;
    const int wn0  = blockIdx.y * 128 + wn * 32;
    const int lr   = lane & 15;           // row (A) / col (B,C) within 16
    const int lh   = lane >> 4;           // half-select

    // Per-lane base pointers for the four A row-chunks and two B col-chunks.
    // A lane layout (ISA 7.12.2, 16-bit A 16x32): lanes 0-15 hold K[8h..8h+8)
    // and K[16+8h..16+8h+8); two contiguous b128 loads per lane.
    const __bf16* aBase[4];
    #pragma unroll
    for (int i = 0; i < 4; ++i)
        aBase[i] = A + (size_t)(wm0 + i * 16 + lr) * K + 8 * lh;
    // B (32x16, 16-bit): lane col = lr, same K order as A: K[8h..8h+8) and K[16+8h..16+8h+8).
    const __bf16* bBase[2];
    #pragma unroll
    for (int j = 0; j < 2; ++j)
        bBase[j] = BT + (size_t)(wn0 + j * 16 + lr) * K + 8 * lh;

    v8f acc[4][2];
    #pragma unroll
    for (int i = 0; i < 4; ++i)
        #pragma unroll
        for (int j = 0; j < 2; ++j)
            #pragma unroll
            for (int e = 0; e < 8; ++e) acc[i][j][e] = 0.0f;

    v16bf a0[4], a1[4], b0[2], b1[2];

    auto loadFrags = [&](int k0, v16bf (&a)[4], v16bf (&b)[2]) {
        #pragma unroll
        for (int i = 0; i < 4; ++i) {
            v8bf lo = *(const v8bf*)(aBase[i] + k0);
            v8bf hi = *(const v8bf*)(aBase[i] + k0 + 16);
            #pragma unroll
            for (int e = 0; e < 8; ++e) { a[i][e] = lo[e]; a[i][8 + e] = hi[e]; }
        }
        #pragma unroll
        for (int j = 0; j < 2; ++j) {
            v8bf lo = *(const v8bf*)(bBase[j] + k0);
            v8bf hi = *(const v8bf*)(bBase[j] + k0 + 16);
            #pragma unroll
            for (int e = 0; e < 8; ++e) { b[j][e] = lo[e]; b[j][8 + e] = hi[e]; }
        }
    };
    auto mmaAll = [&](v16bf (&a)[4], v16bf (&b)[2]) {
        #pragma unroll
        for (int i = 0; i < 4; ++i)
            #pragma unroll
            for (int j = 0; j < 2; ++j)
                acc[i][j] = __builtin_amdgcn_wmma_f32_16x16x32_bf16(
                    false, a[i], false, b[j], (short)0, acc[i][j], false, false);
    };

    loadFrags(0, a0, b0);
    int k0 = 0;
    for (; k0 < K - 64; k0 += 64) {
        // gfx1250 prefetch (global_prefetch_b8) two K-steps ahead on the stream.
        __builtin_prefetch(aBase[0] + k0 + 128, 0, 1);
        __builtin_prefetch(bBase[0] + k0 + 128, 0, 1);
        loadFrags(k0 + 32, a1, b1);   // in flight while first half computes
        mmaAll(a0, b0);
        loadFrags(k0 + 64, a0, b0);   // in flight while second half computes
        mmaAll(a1, b1);
    }
    // tail: k0 == K - 64
    loadFrags(k0 + 32, a1, b1);
    mmaAll(a0, b0);
    mmaAll(a1, b1);

    // Epilogue: C/D layout -> row = base + 8*(lane/16) + r, col = lane%16.
    float* outp = nullptr;
    int rowShift = 0;
    if (MODE == 1) {
        if (wm0 < splitRow) { outp = outA; rowShift = 0; }
        else                { outp = outB; rowShift = splitRow; }
    }
    #pragma unroll
    for (int j = 0; j < 2; ++j) {
        const int col = wn0 + j * 16 + lr;
        const float bv = bias[col];
        #pragma unroll
        for (int i = 0; i < 4; ++i) {
            #pragma unroll
            for (int r = 0; r < 8; ++r) {
                const int row = wm0 + i * 16 + lh * 8 + r;
                float v = acc[i][j][r] + bv;
                v = v > 0.0f ? v : 0.0f;
                if (MODE == 0) outBf[(size_t)row * N + col] = (__bf16)v;
                else           outp[(size_t)(row - rowShift) * N + col] = v;
            }
        }
    }
}

// ---------------------------------------------------------------------------
// y_hat = feat_l @ W3 + b3 : one wave32 per row, shuffle reduction.
// ---------------------------------------------------------------------------
__global__ __launch_bounds__(256) void k_yhat(const float* __restrict__ featL,
                                              const float* __restrict__ W3,
                                              const float* __restrict__ b3,
                                              float* __restrict__ yhat) {
    const int row  = blockIdx.x * 8 + (threadIdx.x >> 5);
    const int lane = threadIdx.x & 31;
    const float* f = featL + (size_t)row * H2DIM;
    float s = 0.0f;
    #pragma unroll
    for (int c = lane; c < H2DIM; c += 32) s += f[c] * W3[c];
    #pragma unroll
    for (int off = 16; off > 0; off >>= 1) s += __shfl_down(s, off, 32);
    if (lane == 0) yhat[row] = s + b3[0];
}

// ---------------------------------------------------------------------------
// Threefry-2x32 (20 rounds) — deterministic stand-in for JAX's pair sampler.
// ---------------------------------------------------------------------------
__device__ __forceinline__ void threefry2x32(unsigned k0, unsigned k1,
                                             unsigned& x0, unsigned& x1) {
    const unsigned R[8] = {13u, 15u, 26u, 6u, 17u, 29u, 16u, 24u};
    unsigned ks[3] = {k0, k1, k0 ^ k1 ^ 0x1BD11BDAu};
    x0 += ks[0]; x1 += ks[1];
    #pragma unroll
    for (int g = 0; g < 5; ++g) {
        #pragma unroll
        for (int r = 0; r < 4; ++r) {
            const unsigned rot = R[(g & 1) * 4 + r];
            x0 += x1;
            x1 = (x1 << rot) | (x1 >> (32 - rot));
            x1 ^= x0;
        }
        x0 += ks[(g + 1) % 3];
        x1 += ks[(g + 2) % 3] + (unsigned)(g + 1);
    }
}

// feat_comb[u] = k*feat_l[a] + (1-k)*feat_l[b], first valid of 8 candidates.
__global__ __launch_bounds__(128) void k_combine(const float* __restrict__ yl,
                                                 const float* __restrict__ yu,
                                                 const float* __restrict__ featL,
                                                 float* __restrict__ featComb) {
    const int u = blockIdx.x;
    int sa = 0, sb = 0, firstA = 0, firstB = 0;
    bool found = false;
    #pragma unroll
    for (int t = 0; t < KTRIES; ++t) {
        unsigned xa0 = (unsigned)u, xa1 = (unsigned)t;
        threefry2x32(42u, 0u, xa0, xa1);
        unsigned xb0 = (unsigned)u, xb1 = (unsigned)t;
        threefry2x32(42u, 1u, xb0, xb1);
        const int ia = (int)(xa0 & (SLN - 1));
        const int ib = (int)(xb0 & (SLN - 1));
        if (t == 0) { firstA = ia; firstB = ib; }
        if (!found && ia != ib && yl[ia] != yl[ib]) { sa = ia; sb = ib; found = true; }
    }
    if (!found) { sa = firstA; sb = firstB; }
    const float ya = yl[sa], yb = yl[sb];
    const float k  = (yu[u] - yb) / (ya - yb);
    const float* fa = featL + (size_t)sa * H2DIM;
    const float* fb = featL + (size_t)sb * H2DIM;
    float* dst = featComb + (size_t)u * H2DIM;
    for (int c = threadIdx.x; c < H2DIM; c += 128)
        dst[c] = k * fa[c] + (1.0f - k) * fb[c];
}

// ---------------------------------------------------------------------------
// Launcher
// ---------------------------------------------------------------------------
extern "C" void kernel_launch(void* const* d_in, const int* in_sizes, int n_in,
                              void* d_out, int out_size, void* d_ws, size_t ws_size,
                              hipStream_t stream) {
    const float* X_l = (const float*)d_in[0];
    const float* y_l = (const float*)d_in[1];
    const float* X_u = (const float*)d_in[2];
    const float* y_u = (const float*)d_in[3];
    const float* W1  = (const float*)d_in[4];
    const float* b1  = (const float*)d_in[5];
    const float* W2  = (const float*)d_in[6];
    const float* b2  = (const float*)d_in[7];
    const float* W3  = (const float*)d_in[8];
    const float* b3  = (const float*)d_in[9];

    float* out      = (float*)d_out;
    float* feat_u   = out;                                   // [SUN, H2]
    float* featComb = out + (size_t)SUN * H2DIM;             // [SUN, H2]
    float* yhat     = out + (size_t)2 * SUN * H2DIM;         // [SLN]

    // Workspace layout (bytes):
    char* ws = (char*)d_ws;
    __bf16* Xbf   = (__bf16*)(ws);                                   // 24576*1024*2 = 50331648
    __bf16* W1T   = (__bf16*)(ws + 50331648);                        // 2048*1024*2  =  4194304
    __bf16* H1bf  = (__bf16*)(ws + 50331648 + 4194304);              // 24576*2048*2 = 100663296
    __bf16* W2T   = (__bf16*)(ws + 50331648 + 4194304 + 100663296);  // 512*2048*2   =  2097152
    float*  featL = (float*)(ws + 50331648 + 4194304 + 100663296 + 2097152); // 8192*512*4

    // 1) precision conversion + weight transpose (one-shot, bandwidth-trivial)
    k_cvt_x <<<(MTOT * DIM_X / 4) / 256, 256, 0, stream>>>(X_l, X_u, Xbf);
    k_cvt_wT<<<(DIM_X * H1DIM) / 256,   256, 0, stream>>>(W1, W1T, DIM_X, H1DIM);
    k_cvt_wT<<<(H1DIM * H2DIM) / 256,   256, 0, stream>>>(W2, W2T, H1DIM, H2DIM);

    // 2) layer 1: H1 = relu(X @ W1 + b1) -> bf16   [24576 x 2048]
    k_gemm_bf16<0><<<dim3(MTOT / 128, H1DIM / 128), 256, 0, stream>>>(
        Xbf, W1T, b1, MTOT, H1DIM, DIM_X, H1bf, nullptr, nullptr, 0);

    // 3) layer 2: F = relu(H1 @ W2 + b2) -> f32, labeled rows to ws, rest to feat_u
    k_gemm_bf16<1><<<dim3(MTOT / 128, H2DIM / 128), 256, 0, stream>>>(
        H1bf, W2T, b2, MTOT, H2DIM, H1DIM, nullptr, featL, feat_u, SLN);

    // 4) y_hat head
    k_yhat<<<SLN / 8, 256, 0, stream>>>(featL, W3, b3, yhat);

    // 5) pair sampling + feature mix
    k_combine<<<SUN, 128, 0, stream>>>(y_l, y_u, featL, featComb);
}
